// Wave_Conv_Self_Attention_38208029065985
// MI455X (gfx1250) — matrix operand order repaired and checked by reading, not compiled, not run on hardware
//
#include <hip/hip_runtime.h>
#include <hip/hip_bf16.h>

typedef __attribute__((ext_vector_type(16))) _Float16 v16h;
typedef __attribute__((ext_vector_type(8)))  _Float16 v8h;
typedef __attribute__((ext_vector_type(8)))  float    v8f;
typedef __attribute__((ext_vector_type(4)))  int      v4i;

#define BATCH 8
#define HI_ 256
#define WI_ 256
#define C_  96
#define NH_ 3
#define DH_ 32
#define BW_ 8192      // 8 * 32 * 32 windows
#define NWIN 64       // 8x8 window
#define HB 128        // half-res band image edge
#define LOG_MAX_ 4.6051701859880914f

static __device__ __forceinline__ float leaky02(float v) { return v > 0.f ? v : 0.2f * v; }

// ---------------- WMMA fragment loaders (CDNA5 ISA 7.12.2, wave32) ----------------
// A matrix 16x32 f16. Per lane: halves 0..7 = K kb..kb+7, halves 8..15 = K kb+16..kb+23
// (kb = 8 for lanes 16..31). Two contiguous 16-byte loads + shuffle.
__device__ __forceinline__ v16h load_a16x32(const _Float16* A, int lda, int lane) {
  const _Float16* row = A + (long long)(lane & 15) * lda + ((lane & 16) ? 8 : 0);
  v8h lo = *(const v8h*)(row);
  v8h hi = *(const v8h*)(row + 16);
  return __builtin_shufflevector(lo, hi, 0, 1, 2, 3, 4, 5, 6, 7, 8, 9, 10, 11, 12, 13, 14, 15);
}
// B matrix 32x16 f16 from an N-major (N x K) source: same K order as A frags
// (halves 0..7 = K kb..kb+7, 8..15 = K kb+16..kb+23, kb = 8 for lanes 16..31), N = lane&15.
__device__ __forceinline__ v16h load_bt(const _Float16* Bt, int ldk, int lane) {
  const _Float16* row = Bt + (long long)(lane & 15) * ldk + ((lane & 16) ? 8 : 0);
  v8h lo = *(const v8h*)(row);
  v8h hi = *(const v8h*)(row + 16);
  return __builtin_shufflevector(lo, hi, 0, 1, 2, 3, 4, 5, 6, 7, 8, 9, 10, 11, 12, 13, 14, 15);
}

// ---------------- generic batched WMMA GEMM: O = leaky?(A@Bt^T + bias) ----------------
// A: per-batch (M x K) f16 row-major; Bt: shared weights stored N-major (N x K) f16.
__global__ void wmma_gemm(const _Float16* __restrict__ A, long long sA,
                          const _Float16* __restrict__ Bt,
                          const float* __restrict__ bias,
                          _Float16* __restrict__ Oh, float* __restrict__ Of,
                          long long sO, int ldo, int coff,
                          int M, int K, int N, int doLeaky) {
  const int lane = threadIdx.x & 31;
  const int wave = threadIdx.x >> 5;
  const int tile = blockIdx.x * 4 + wave;
  const int tn = N >> 4;
  const int tiles = (M >> 4) * tn;
  if (tile >= tiles) return;
  const int mt = tile / tn, nt = tile % tn;
  const _Float16* Ab  = A + (long long)blockIdx.y * sA + (long long)(mt * 16) * K;
  const _Float16* Btb = Bt + (long long)(nt * 16) * K;
  __builtin_prefetch(Btb + (long long)(lane & 15) * K, 0, 3);
  v8f c = {};
  for (int k0 = 0; k0 < K; k0 += 32) {
    v16h a = load_a16x32(Ab + k0, K, lane);
    v16h b = load_bt(Btb + k0, K, lane);
    c = __builtin_amdgcn_wmma_f32_16x16x32_f16(false, a, false, b, (short)0, c, false, false);
  }
  const int n = nt * 16 + (lane & 15);
  const float bs = bias ? bias[n] : 0.f;
  const long long obase = (long long)blockIdx.y * sO + coff + n;
#pragma unroll
  for (int v = 0; v < 8; ++v) {
    const int m = mt * 16 + v + ((lane & 16) ? 8 : 0);
    float val = c[v] + bs;
    if (doLeaky) val = leaky02(val);
    if (Oh) Oh[obase + (long long)m * ldo] = (_Float16)val;
    if (Of) Of[obase + (long long)m * ldo] = val;
  }
}

// ---------------- casts ----------------
__global__ void cast_f32_to_f16(const float* __restrict__ s, _Float16* __restrict__ d, long long n) {
  long long i = blockIdx.x * (long long)blockDim.x + threadIdx.x;
  if (i < n) d[i] = (_Float16)s[i];
}
// transpose-cast: src (K x N) f32 row-major -> dst (N x K) f16 (N-major weights for B frags)
__global__ void cast_transpose(const float* __restrict__ s, _Float16* __restrict__ d, int K, int N) {
  const int i = blockIdx.x * blockDim.x + threadIdx.x;
  if (i >= K * N) return;
  const int k = i / N, n = i % N;
  d[n * K + k] = (_Float16)s[i];
}

// ---------------- Haar DWT: per (window, channel) ----------------
__global__ void dwt_kernel(const float* __restrict__ x,
                           _Float16* __restrict__ ll16,
                           _Float16* __restrict__ hcat16,
                           float* __restrict__ bands) {
  const int tid = blockIdx.x * blockDim.x + threadIdx.x;
  if (tid >= BW_ * C_) return;
  const int w = tid / C_, c = tid % C_;
  const int b = w >> 10, wy = (w >> 5) & 31, wx = w & 31;
  const float* xw = x + (long long)w * NWIN * C_ + c;
  _Float16* llw = ll16 + (long long)w * 16 * C_ + c;
  _Float16* hw  = hcat16 + (long long)w * 16 * 3 * C_ + c;
  const long long BSZ = (long long)BATCH * C_ * HB * HB;
  const long long ibase = (long long)(b * C_ + c) * HB * HB;
#pragma unroll
  for (int i = 0; i < 4; ++i)
#pragma unroll
    for (int j = 0; j < 4; ++j) {
      const float a  = xw[((2 * i) * 8 + 2 * j) * C_];
      const float bb = xw[((2 * i) * 8 + 2 * j + 1) * C_];
      const float cc = xw[((2 * i + 1) * 8 + 2 * j) * C_];
      const float dd = xw[((2 * i + 1) * 8 + 2 * j + 1) * C_];
      const float ll = (a + bb + cc + dd) * 0.5f;
      const float lh = (bb + dd - a - cc) * 0.5f;
      const float hl = (cc + dd - a - bb) * 0.5f;
      const float hh = (a + dd - bb - cc) * 0.5f;
      const int p = i * 4 + j;
      llw[p * C_] = (_Float16)ll;
      hw[p * 3 * C_]          = (_Float16)lh;
      hw[p * 3 * C_ + C_]     = (_Float16)hl;
      hw[p * 3 * C_ + 2 * C_] = (_Float16)hh;
      const long long pix = ibase + (long long)(wy * 4 + i) * HB + (wx * 4 + j);
      bands[pix]           = ll;
      bands[BSZ + pix]     = lh;
      bands[2 * BSZ + pix] = hl;
      bands[3 * BSZ + pix] = hh;
    }
}

// ---------------- depthwise 3x3 + leaky on each band image ----------------
__global__ void bandconv_kernel(const float* __restrict__ bands,
                                const float* __restrict__ wl, const float* __restrict__ bl,
                                const float* __restrict__ wh, const float* __restrict__ bh,
                                float* __restrict__ out) {
  const long long BSZ = (long long)BATCH * C_ * HB * HB;
  long long tid = blockIdx.x * (long long)blockDim.x + threadIdx.x;
  if (tid >= 4 * BSZ) return;
  const int band = (int)(tid / BSZ);
  const long long r = tid % BSZ;
  const int xp = (int)(r % HB);
  const int yp = (int)((r / HB) % HB);
  const int bc = (int)(r / (HB * HB));
  const int c = bc % C_;
  const float* wgt = ((band == 0) ? wl : wh) + c * 9;
  float acc = (band == 0) ? bl[c] : bh[c];
  const float* img = bands + band * BSZ + (long long)bc * HB * HB;
#pragma unroll
  for (int dy = -1; dy <= 1; ++dy)
#pragma unroll
    for (int dx = -1; dx <= 1; ++dx) {
      const int Y = yp + dy, X = xp + dx;
      if (Y < 0 || Y >= HB || X < 0 || X >= HB) continue;
      acc += img[(long long)Y * HB + X] * wgt[(dy + 1) * 3 + (dx + 1)];
    }
  out[tid] = leaky02(acc);
}

// ---------------- fused iDWT + depthwise 3x3 (rconv) + window-partition ----------------
__global__ void rst_kernel(const float* __restrict__ bandsc,
                           const float* __restrict__ w_r, const float* __restrict__ b_r,
                           _Float16* __restrict__ concat16) {
  const long long TOT = (long long)BATCH * C_ * HI_ * WI_;
  long long tid = blockIdx.x * (long long)blockDim.x + threadIdx.x;
  if (tid >= TOT) return;
  const int xp = (int)(tid % WI_);
  long long t = tid / WI_;
  const int yp = (int)(t % HI_); t /= HI_;
  const int c = (int)(t % C_);
  const int b = (int)(t / C_);
  const long long BSZ = (long long)BATCH * C_ * HB * HB;
  const float* llc = bandsc;
  const float* lhc = bandsc + BSZ;
  const float* hlc = bandsc + 2 * BSZ;
  const float* hhc = bandsc + 3 * BSZ;
  const long long cbase = (long long)(b * C_ + c) * HB * HB;
  float acc = b_r[c];
#pragma unroll
  for (int dy = -1; dy <= 1; ++dy)
#pragma unroll
    for (int dx = -1; dx <= 1; ++dx) {
      const int Y = yp + dy, X = xp + dx;
      if (Y < 0 || Y >= HI_ || X < 0 || X >= WI_) continue;
      const int i = Y >> 1, j = X >> 1, di = Y & 1, dj = X & 1;
      const long long pix = cbase + (long long)i * HB + j;
      const float ll = llc[pix], lh = lhc[pix], hl = hlc[pix], hh = hhc[pix];
      const float slh = dj ? 1.f : -1.f;
      const float shl = di ? 1.f : -1.f;
      const float shh = (di == dj) ? 1.f : -1.f;
      acc += 0.5f * (ll + slh * lh + shl * hl + shh * hh) * w_r[c * 9 + (dy + 1) * 3 + (dx + 1)];
    }
  const float val = leaky02(acc);
  const int ww = b * 1024 + (yp >> 3) * 32 + (xp >> 3);
  const int n = (yp & 7) * 8 + (xp & 7);
  concat16[(long long)ww * NWIN * (3 * C_) + (long long)n * (3 * C_) + 2 * C_ + c] = (_Float16)val;
}

// ---------------- per (window, head, branch) attention with WMMA ----------------
__global__ void attn_kernel(const _Float16* __restrict__ Q16,
                            const _Float16* __restrict__ kvl,
                            const _Float16* __restrict__ kvh,
                            const float* __restrict__ ls_low,
                            const float* __restrict__ ls_high,
                            _Float16* __restrict__ concat16) {
  __shared__ _Float16 sKV[16 * 192]; // raw kv block (async staged)
  __shared__ _Float16 sQ[64 * 32];   // normalized Q head (row-major M x K)
  __shared__ _Float16 sK[16 * 32];   // normalized K rows (N-major for B frag)
  __shared__ _Float16 sVt[32 * 32];  // V transposed (d-major: sVt[d][k]), k 16..23 zero
  __shared__ float    sS[64 * 16];   // scaled logits
  __shared__ _Float16 sA[64 * 32];   // softmax probs, cols 16..23 zero
  const int lane = threadIdx.x;
  const int w = blockIdx.x;
  const int branch = blockIdx.y / NH_;
  const int h = blockIdx.y % NH_;
  const _Float16* kv = (branch ? kvh : kvl) + (long long)w * 16 * 2 * C_;
  const float ls = branch ? ls_high[h] : ls_low[h];
  const float scale = __expf(fminf(ls, LOG_MAX_));

  // ---- stage kv block (6144 B) into LDS via CDNA5 async-to-LDS path ----
#if __has_builtin(__builtin_amdgcn_global_load_async_to_lds_b128)
  {
    const char* g = (const char*)kv;
    char* l = (char*)sKV;
#pragma unroll
    for (int it = 0; it < 12; ++it) {
      const int off = (it * 32 + lane) * 16;
      __builtin_amdgcn_global_load_async_to_lds_b128(
          (__attribute__((address_space(1))) v4i*)(g + off),
          (__attribute__((address_space(3))) v4i*)(l + off), 0, 0);
    }
    asm volatile("s_wait_asynccnt 0x0" ::: "memory");
  }
#else
  for (int idx = lane; idx < 16 * 192 / 8; idx += 32)
    ((v8h*)sKV)[idx] = ((const v8h*)kv)[idx];
#endif

  // ---- l2-normalize Q rows (2 rows per lane), vectorized ----
  const _Float16* Qw = Q16 + (long long)w * NWIN * C_ + h * DH_;
#pragma unroll
  for (int rr = 0; rr < 2; ++rr) {
    const int r = lane * 2 + rr;
    const _Float16* qp = Qw + (long long)r * C_;
    v16h q0 = *(const v16h*)(qp);
    v16h q1 = *(const v16h*)(qp + 16);
    float ss = 0.f;
#pragma unroll
    for (int i = 0; i < 16; ++i) { float f0 = (float)q0[i], f1 = (float)q1[i]; ss += f0 * f0 + f1 * f1; }
    const float rn = 1.f / fmaxf(sqrtf(ss), 1e-12f);
    v16h o0, o1;
#pragma unroll
    for (int i = 0; i < 16; ++i) { o0[i] = (_Float16)((float)q0[i] * rn); o1[i] = (_Float16)((float)q1[i] * rn); }
    *(v16h*)(sQ + r * DH_)      = o0;
    *(v16h*)(sQ + r * DH_ + 16) = o1;
  }
  __syncthreads();

  // ---- K normalize (lanes 0..15); V transpose + zero-pad (all lanes) ----
  {
    const int d = lane;            // each lane owns one output dim d of V^T
    v8h z = {};
    *(v8h*)(sVt + d * 32 + 16) = z;  // zero pad k = 16..23
#pragma unroll
    for (int p = 0; p < 16; ++p)
      sVt[d * 32 + p] = sKV[p * (2 * C_) + C_ + h * DH_ + d];
  }
  if (lane < 16) {
    const _Float16* kp = sKV + lane * (2 * C_) + h * DH_;
    v16h k0 = *(const v16h*)(kp);
    v16h k1 = *(const v16h*)(kp + 16);
    float ss = 0.f;
#pragma unroll
    for (int i = 0; i < 16; ++i) { float f0 = (float)k0[i], f1 = (float)k1[i]; ss += f0 * f0 + f1 * f1; }
    const float rn = 1.f / fmaxf(sqrtf(ss), 1e-12f);
    v16h o0, o1;
#pragma unroll
    for (int i = 0; i < 16; ++i) { o0[i] = (_Float16)((float)k0[i] * rn); o1[i] = (_Float16)((float)k1[i] * rn); }
    *(v16h*)(sK + lane * DH_)      = o0;
    *(v16h*)(sK + lane * DH_ + 16) = o1;
  }
  __syncthreads();

  // ---- S = Qn @ Kn^T (64x16): B frag = two 16-byte LDS loads (sK is N-major) ----
  {
    const v16h b = load_bt(sK, DH_, lane);
#pragma unroll
    for (int mt = 0; mt < 4; ++mt) {
      v16h a = load_a16x32(sQ + mt * 16 * DH_, DH_, lane);
      v8f cacc = {};
      cacc = __builtin_amdgcn_wmma_f32_16x16x32_f16(false, a, false, b, (short)0, cacc, false, false);
      const int n = lane & 15;
#pragma unroll
      for (int v = 0; v < 8; ++v) {
        const int m = mt * 16 + v + ((lane & 16) ? 8 : 0);
        sS[m * 16 + n] = cacc[v] * scale;
      }
    }
  }
  __syncthreads();

  // ---- softmax over 16 keys (2 rows per lane); vectorized prob store ----
#pragma unroll
  for (int rr = 0; rr < 2; ++rr) {
    const int r = lane * 2 + rr;
    float mx = -1e30f;
#pragma unroll
    for (int j = 0; j < 16; ++j) mx = fmaxf(mx, sS[r * 16 + j]);
    float sum = 0.f, e[16];
#pragma unroll
    for (int j = 0; j < 16; ++j) { e[j] = __expf(sS[r * 16 + j] - mx); sum += e[j]; }
    const float inv = 1.f / sum;
    v16h pr;
#pragma unroll
    for (int j = 0; j < 16; ++j) pr[j] = (_Float16)(e[j] * inv);
    *(v16h*)(sA + r * 32) = pr;
    v8h z = {};
    *(v8h*)(sA + r * 32 + 16) = z;  // zero pad K dim 16..23
  }
  __syncthreads();

  // ---- O = A @ V (64x32): B frag from sVt (d-major) = two 16-byte LDS loads ----
  _Float16* outw = concat16 + (long long)w * NWIN * (3 * C_) + branch * C_;
#pragma unroll
  for (int nt = 0; nt < 2; ++nt) {
    const v16h b = load_bt(sVt + nt * 16 * 32, 32, lane);
#pragma unroll
    for (int mt = 0; mt < 4; ++mt) {
      v16h a = load_a16x32(sA + mt * 16 * 32, 32, lane);
      v8f cacc = {};
      cacc = __builtin_amdgcn_wmma_f32_16x16x32_f16(false, a, false, b, (short)0, cacc, false, false);
      const int d = nt * 16 + (lane & 15);
      const int col = d * NH_ + h;   // channel interleave: c = d*NH + h
#pragma unroll
      for (int v = 0; v < 8; ++v) {
        const int m = mt * 16 + v + ((lane & 16) ? 8 : 0);
        outw[(long long)m * (3 * C_) + col] = (_Float16)cacc[v];
      }
    }
  }
}

extern "C" void kernel_launch(void* const* d_in, const int* in_sizes, int n_in,
                              void* d_out, int out_size, void* d_ws, size_t ws_size,
                              hipStream_t stream) {
  (void)in_sizes; (void)n_in; (void)out_size; (void)ws_size;
  const float* x     = (const float*)d_in[0];
  const float* w_q   = (const float*)d_in[1];
  const float* b_q   = (const float*)d_in[2];
  const float* w_kvl = (const float*)d_in[3];
  const float* b_kvl = (const float*)d_in[4];
  const float* w_kvh = (const float*)d_in[5];
  const float* b_kvh = (const float*)d_in[6];
  const float* lsl   = (const float*)d_in[7];
  const float* lsh   = (const float*)d_in[8];
  const float* w_lc  = (const float*)d_in[9];
  const float* b_lc  = (const float*)d_in[10];
  const float* w_hc  = (const float*)d_in[11];
  const float* b_hc  = (const float*)d_in[12];
  const float* w_rc  = (const float*)d_in[13];
  const float* b_rc  = (const float*)d_in[14];
  const float* w_fh  = (const float*)d_in[15];
  const float* b_fh  = (const float*)d_in[16];
  const float* w_fa  = (const float*)d_in[17];
  const float* b_fa  = (const float*)d_in[18];
  const float* w_p   = (const float*)d_in[19];
  const float* b_p   = (const float*)d_in[20];

  char* ws = (char*)d_ws;
  auto alloc16 = [&](long long nelem) { _Float16* p = (_Float16*)ws; ws += ((nelem * 2 + 255) & ~255LL); return p; };
  auto alloc32 = [&](long long nelem) { float* p = (float*)ws; ws += ((nelem * 4 + 255) & ~255LL); return p; };

  const long long NX = (long long)BW_ * NWIN * C_;
  _Float16* x16    = alloc16(NX);
  _Float16* q16    = alloc16(NX);
  _Float16* ll16   = alloc16((long long)BW_ * 16 * C_);
  _Float16* hc16   = alloc16((long long)BW_ * 16 * 3 * C_);
  _Float16* xh16   = alloc16((long long)BW_ * 16 * C_);
  _Float16* kvl16  = alloc16((long long)BW_ * 16 * 2 * C_);
  _Float16* kvh16  = alloc16((long long)BW_ * 16 * 2 * C_);
  _Float16* cat16  = alloc16((long long)BW_ * NWIN * 3 * C_);
  _Float16* tmp16  = alloc16(NX);
  _Float16* wq16   = alloc16(96 * 96);     // stored N-major (N x K)
  _Float16* wkvl16 = alloc16(192 * 96);
  _Float16* wkvh16 = alloc16(192 * 96);
  _Float16* wfh16  = alloc16(96 * 288);    // w_fh already N-major (O x I)
  _Float16* wfa16  = alloc16(96 * 288);
  _Float16* wp16   = alloc16(96 * 96);
  float* bands  = alloc32(4LL * BATCH * C_ * HB * HB);
  float* bandsc = alloc32(4LL * BATCH * C_ * HB * HB);

  const int T = 256;
  cast_f32_to_f16<<<(int)((NX + T - 1) / T), T, 0, stream>>>(x, x16, NX);
  cast_transpose<<<(9216  + T - 1) / T, T, 0, stream>>>(w_q,   wq16,   96, 96);
  cast_transpose<<<(18432 + T - 1) / T, T, 0, stream>>>(w_kvl, wkvl16, 96, 192);
  cast_transpose<<<(18432 + T - 1) / T, T, 0, stream>>>(w_kvh, wkvh16, 96, 192);
  cast_transpose<<<(27648 + T - 1) / T, T, 0, stream>>>(w_fa,  wfa16,  288, 96);
  cast_transpose<<<(9216  + T - 1) / T, T, 0, stream>>>(w_p,   wp16,   96, 96);
  cast_f32_to_f16<<<(27648 + T - 1) / T, T, 0, stream>>>(w_fh, wfh16, 27648);

  dwt_kernel<<<(BW_ * C_ + T - 1) / T, T, 0, stream>>>(x, ll16, hc16, bands);

  auto gemm = [&](const _Float16* A, long long sA, const _Float16* B, const float* bias,
                  _Float16* Oh, float* Of, long long sO, int ldo, int coff,
                  int M, int K, int N, int leaky) {
    const int tiles = (M / 16) * (N / 16);
    dim3 grid((tiles + 3) / 4, BW_);
    wmma_gemm<<<grid, 128, 0, stream>>>(A, sA, B, bias, Oh, Of, sO, ldo, coff, M, K, N, leaky);
  };
  gemm(x16,  64 * 96,  wq16,   b_q,   q16,   nullptr, 64 * 96,  96,  0, 64, 96,  96,  0);  // Q
  gemm(ll16, 16 * 96,  wkvl16, b_kvl, kvl16, nullptr, 16 * 192, 192, 0, 16, 96,  192, 0);  // kv_l
  gemm(hc16, 16 * 288, wfh16,  b_fh,  xh16,  nullptr, 16 * 96,  96,  0, 16, 288, 96,  1);  // x_h (leaky)
  gemm(xh16, 16 * 96,  wkvh16, b_kvh, kvh16, nullptr, 16 * 192, 192, 0, 16, 96,  192, 0);  // kv_h

  const long long nbc = 4LL * BATCH * C_ * HB * HB;
  bandconv_kernel<<<(int)((nbc + T - 1) / T), T, 0, stream>>>(bands, w_lc, b_lc, w_hc, b_hc, bandsc);

  attn_kernel<<<dim3(BW_, NH_ * 2), 32, 0, stream>>>(q16, kvl16, kvh16, lsl, lsh, cat16);

  const long long nrst = (long long)BATCH * C_ * HI_ * WI_;
  rst_kernel<<<(int)((nrst + T - 1) / T), T, 0, stream>>>(bandsc, w_rc, b_rc, cat16);

  gemm(cat16, 64 * 288, wfa16, b_fa, tmp16, nullptr,         64 * 96, 96, 0, 64, 288, 96, 0); // fa
  gemm(tmp16, 64 * 96,  wp16,  b_p,  nullptr, (float*)d_out, 64 * 96, 96, 0, 64, 96,  96, 0); // p -> f32 out
}
